// AdaptiveGatedAttentionFusion_52613349376381
// MI455X (gfx1250) — compile-verified, matrix operand order checked by reading, not run on hardware
//
#include <hip/hip_runtime.h>
#include <math.h>

// ---------------------------------------------------------------------------
// MI455X (gfx1250) implementation.
// Compute-bound (~300 GFLOP vs ~150MB weights < 192MB L2): all big GEMMs run
// as bf16 WMMA (v_wmma_f32_16x16x32_bf16, f32 accum). wave32 everywhere.
// GEMM stages A/B tiles to LDS with async copies (global_load_async_to_lds,
// ASYNCcnt) and double-buffers them; 8 waves consume each staged tile.
// Activations kept spatial-major [b][s][c] so conv1x1 operands are
// K-contiguous for WMMA fragment loads; Q/K stored channel-major via a
// transposed GEMM epilogue so the spatial-contraction score GEMM is also
// K-contiguous. Residuals + biases are GEMM epilogues.
// ---------------------------------------------------------------------------

typedef __bf16 bf16;
typedef __bf16 v16bf __attribute__((ext_vector_type(16)));
typedef float  v8f   __attribute__((ext_vector_type(8)));
typedef float  f32x4 __attribute__((ext_vector_type(4)));

union FragU { v16bf v; f32x4 q[2]; };

#define BATCH 4
#define HWSP 1024       // H*W = 32*32
#define BCH 1024        // balanced channels
#define NHEADS 8
#define HDIM 128

// LDS tile: 128 rows x 32 k-elems (64B) padded to 80B stride (conflict-free
// b128 fragment reads across the 16-lane groups). One tile = 10240B.
#define TROWB 80
#define TILEB 10240

#define ASYNC_B128(ldsaddr, gptr)                                   \
  asm volatile("global_load_async_to_lds_b128 %0, %1, off"          \
               :: "v"(ldsaddr), "v"(gptr) : "memory")

__device__ __forceinline__ float wave_sum(float v) {
#pragma unroll
  for (int m = 16; m >= 1; m >>= 1) v += __shfl_xor(v, m, 32);
  return v;
}
__device__ __forceinline__ float wave_max(float v) {
#pragma unroll
  for (int m = 16; m >= 1; m >>= 1) v = fmaxf(v, __shfl_xor(v, m, 32));
  return v;
}

// ---------------------------------------------------------------------------
// Generic batched WMMA GEMM:  C[m,n] = sum_k A[m,k]*B[n,k]  (+bias[n]) (+=)
// Both operands row-major over K (K-contiguous), matching the CDNA5 WMMA
// fragment layout (A: lane=M row; B: lane=N column => D = A * B^T).
// Block = 256 threads = 8 waves; block tile 128x128; wave tile 64x32
// (4x2 v_wmma_f32_16x16x32_bf16 tiles). A/B 128x32 tiles double-buffered in
// LDS via async copies. M,N multiples of 128; K multiple of 32.
// z = b*heads + h selects batch/head via strides.
// ---------------------------------------------------------------------------
template <bool OUT_BF16, bool TRANS, bool HAS_BIAS, bool RESID>
__global__ __launch_bounds__(256) void wmma_gemm(
    const bf16* __restrict__ A, const bf16* __restrict__ B,
    const float* __restrict__ bias, void* __restrict__ C,
    int K, int lda, int ldb, int ldo,
    long long sAb, long long sAh, long long sBb, long long sBh,
    long long sCb, long long sCh, int heads) {
  __shared__ __align__(16) char smem[4 * TILEB];  // A0 A1 B0 B1

  const int z = blockIdx.z;
  const int b = z / heads, h = z % heads;
  A += (long long)b * sAb + (long long)h * sAh;
  B += (long long)b * sBb + (long long)h * sBh;
  const long long coff = (long long)b * sCb + (long long)h * sCh;

  const int tid = threadIdx.x;
  const int lane = tid & 31;
  const int wave = tid >> 5;
  const int wm = wave >> 2;             // 0..1 -> 64-row slice
  const int wn = wave & 3;              // 0..3 -> 32-col slice
  const int mBase = blockIdx.y * 128;
  const int nBase = blockIdx.x * 128;

  const uint32_t lds0 = (uint32_t)(uintptr_t)(void*)smem;  // LDS byte offset

  // async staging: thread loads 32B (2 x b128) of one row of A and of B
  const int srow = tid >> 1;            // 0..127
  const int qe = (tid & 1) * 16;        // element offset within the 32-k row

  auto issue = [&](int t) {
    const int p = t & 1;
    const int k = t * 32;
    const bf16* ga = A + (long long)(mBase + srow) * lda + k + qe;
    const bf16* gb = B + (long long)(nBase + srow) * ldb + k + qe;
    const uint32_t la = lds0 + p * TILEB + srow * TROWB + qe * 2;
    const uint32_t lb = lds0 + 2 * TILEB + p * TILEB + srow * TROWB + qe * 2;
    ASYNC_B128(la, ga);
    ASYNC_B128(la + 16, ga + 8);
    ASYNC_B128(lb, gb);
    ASYNC_B128(lb + 16, gb + 8);
    if (k + 64 < K) __builtin_prefetch(ga + 64, 0, 3);
  };

  v8f acc[4][2] = {};
  // fragment lane layout: lanes 0-15 hold K [0,8)u[16,24), 16-31 [8,16)u[24,32)
  const int kh2 = (lane >> 4) * 16;     // byte offset of first 16B chunk
  const int fr = lane & 15;
  const int T = K >> 5;

  issue(0);
  for (int t = 0; t < T; ++t) {
    const int p = t & 1;
    asm volatile("s_wait_asynccnt 0x0" ::: "memory");  // my tile-t copies done
    __syncthreads();                                   // everyone's tile ready
    if (t + 1 < T) issue(t + 1);                       // prefetch next buffer

    const char* sa = smem + p * TILEB;
    const char* sb = smem + 2 * TILEB + p * TILEB;
    FragU fa[4], fb[2];
#pragma unroll
    for (int i = 0; i < 4; ++i) {
      const char* pa = sa + (wm * 64 + 16 * i + fr) * TROWB + kh2;
      fa[i].q[0] = *(const f32x4*)pa;
      fa[i].q[1] = *(const f32x4*)(pa + 32);
    }
#pragma unroll
    for (int j = 0; j < 2; ++j) {
      const char* pb = sb + (wn * 32 + 16 * j + fr) * TROWB + kh2;
      fb[j].q[0] = *(const f32x4*)pb;
      fb[j].q[1] = *(const f32x4*)(pb + 32);
    }
#pragma unroll
    for (int i = 0; i < 4; ++i)
#pragma unroll
      for (int j = 0; j < 2; ++j)
        acc[i][j] = __builtin_amdgcn_wmma_f32_16x16x32_bf16(
            false, fa[i].v, false, fb[j].v, (short)0, acc[i][j], false, false);
  }

  float* Cf = (float*)C;
  bf16* Cb = (bf16*)C;
  const int mlane = (lane >> 4) * 8;    // C layout: VGPR r -> M=r (+8 upper)
#pragma unroll
  for (int i = 0; i < 4; ++i) {
#pragma unroll
    for (int j = 0; j < 2; ++j) {
      const int n = nBase + wn * 32 + 16 * j + (lane & 15);
      float bv = 0.0f;
      if constexpr (HAS_BIAS) bv = bias[n];
#pragma unroll
      for (int r = 0; r < 8; ++r) {
        const int m = mBase + wm * 64 + 16 * i + mlane + r;
        const float v = acc[i][j][r] + bv;
        const long long idx = TRANS ? coff + (long long)n * ldo + m
                                    : coff + (long long)m * ldo + n;
        if constexpr (OUT_BF16) {
          Cb[idx] = (bf16)v;
        } else if constexpr (RESID) {
          Cf[idx] += v;
        } else {
          Cf[idx] = v;
        }
      }
    }
  }
}

// f32 [B,C,HW] channel-major -> bf16 [B,HW,C] spatial-major
__global__ void to_sm_bf16(const float* __restrict__ in, bf16* __restrict__ out,
                           int C, int total) {
  int t = blockIdx.x * blockDim.x + threadIdx.x;
  if (t >= total) return;
  int c = t % C;
  int s = (t / C) & (HWSP - 1);
  int b = t / (C * HWSP);
  out[t] = (bf16)in[((long long)b * C + c) * HWSP + s];
}

__global__ void to_bf16(const float* __restrict__ in, bf16* __restrict__ out, int n) {
  int t = blockIdx.x * blockDim.x + threadIdx.x;
  if (t < n) out[t] = (bf16)in[t];
}

// Per (b,s) row: L2-normalize over 1024 channels; write bf16 (GEMM operand)
// and f32 (residual init for the cross buffer); compute quality metric.
// mode 0: unbiased variance over channels; mode 1: zero-fraction.
__global__ __launch_bounds__(256) void l2norm_kernel(
    const float* __restrict__ raw, bf16* __restrict__ norm_bf,
    float* __restrict__ cross_init, float* __restrict__ metric, int mode) {
  const int lane = threadIdx.x & 31;
  const int row = blockIdx.x * 8 + (threadIdx.x >> 5);  // b*HW + s
  const float* x = raw + (long long)row * BCH;
  float sum = 0.f, ss = 0.f, zc = 0.f;
  for (int c = lane; c < BCH; c += 32) {
    float v = x[c];
    sum += v; ss += v * v; zc += (v == 0.0f) ? 1.0f : 0.0f;
  }
  sum = wave_sum(sum); ss = wave_sum(ss); zc = wave_sum(zc);
  const float inv = 1.0f / fmaxf(sqrtf(ss), 1e-12f);
  bf16* nb = norm_bf + (long long)row * BCH;
  float* ci = cross_init + (long long)row * BCH;
  for (int c = lane; c < BCH; c += 32) {
    float y = x[c] * inv;
    nb[c] = (bf16)y;
    ci[c] = y;
  }
  if (lane == 0) {
    if (mode == 0) {
      float sy = sum * inv, ssy = ss * inv * inv;
      metric[row] = (ssy - sy * sy / (float)BCH) / (float)(BCH - 1);
    } else {
      metric[row] = zc / (float)BCH;
    }
  }
}

// scores f32 [B*H*128, 128] -> softmax(scale*s) -> bf16 attn
__global__ __launch_bounds__(256) void softmax_kernel(
    const float* __restrict__ scores, bf16* __restrict__ attn) {
  const int lane = threadIdx.x & 31;
  const int row = blockIdx.x * 8 + (threadIdx.x >> 5);
  const float scale = 0.08838834764831845f;  // 128^-0.5
  const float* s = scores + (long long)row * HDIM;
  float v[4];
  float mx = -3.0e38f;
#pragma unroll
  for (int i = 0; i < 4; ++i) { v[i] = s[lane + 32 * i] * scale; mx = fmaxf(mx, v[i]); }
  mx = wave_max(mx);
  float sm = 0.f;
#pragma unroll
  for (int i = 0; i < 4; ++i) { v[i] = __expf(v[i] - mx); sm += v[i]; }
  sm = wave_sum(sm);
  const float inv = 1.0f / sm;
  bf16* a = attn + (long long)row * HDIM;
#pragma unroll
  for (int i = 0; i < 4; ++i) a[lane + 32 * i] = (bf16)(v[i] * inv);
}

// mean over spatial: cross [b][s][c] -> pool [b][c]
__global__ void pool_kernel(const float* __restrict__ cross, float* __restrict__ pool) {
  int t = blockIdx.x * blockDim.x + threadIdx.x;
  if (t >= BATCH * BCH) return;
  int b = t / BCH, c = t % BCH;
  const float* p = cross + (long long)b * HWSP * BCH + c;
  float s = 0.f;
  for (int i = 0; i < HWSP; ++i) s += p[(long long)i * BCH];
  pool[t] = s * (1.0f / HWSP);
}

// SE: h = relu(W1 p + b1)[64]; se = sigmoid(W2 h + b2)[1024]. One block per b.
__global__ __launch_bounds__(256) void se_kernel(
    const float* __restrict__ pool, const float* __restrict__ w1,
    const float* __restrict__ b1, const float* __restrict__ w2,
    const float* __restrict__ b2, float* __restrict__ se) {
  __shared__ float p[BCH];
  __shared__ float h[64];
  const int b = blockIdx.x, tid = threadIdx.x;
  for (int c = tid; c < BCH; c += 256) p[c] = pool[b * BCH + c];
  __syncthreads();
  if (tid < 64) {
    float acc = b1[tid];
    const float* w = w1 + tid * BCH;
    for (int c = 0; c < BCH; ++c) acc += w[c] * p[c];
    h[tid] = fmaxf(acc, 0.0f);
  }
  __syncthreads();
  for (int c = tid; c < BCH; c += 256) {
    float acc = b2[c];
    const float* w = w2 + c * 64;
#pragma unroll
    for (int j = 0; j < 64; ++j) acc += w[j] * h[j];
    se[b * BCH + c] = 1.0f / (1.0f + __expf(-acc));
  }
}

// Per (b,s) row: gates = sigmoid(Wg . [rgb_ref, depth_ref, lidar_ref, q0..2]);
// fused[s][0:3072] = [rgb_ref*g0, depth_ref*g1, lidar_ref*g2] in bf16.
__global__ __launch_bounds__(256) void gate_fused_kernel(
    const float* __restrict__ cross_r, const float* __restrict__ cross_d,
    const float* __restrict__ cross_l, const float* __restrict__ se_r,
    const float* __restrict__ se_d, const float* __restrict__ se_l,
    const float* __restrict__ var_r, const float* __restrict__ sp_d,
    const float* __restrict__ sp_l, const float* __restrict__ gw,
    const float* __restrict__ gb, bf16* __restrict__ fused) {
  const int lane = threadIdx.x & 31;
  const int row = blockIdx.x * 8 + (threadIdx.x >> 5);  // b*HW + s
  const int b = row >> 10;
  const long long base = (long long)row * BCH;
  const float* sr = se_r + b * BCH;
  const float* sd = se_d + b * BCH;
  const float* sl = se_l + b * BCH;
  float a0 = 0.f, a1 = 0.f, a2 = 0.f;
  for (int c = lane; c < BCH; c += 32) {
    float vr = cross_r[base + c] * sr[c];
    float vd = cross_d[base + c] * sd[c];
    float vl = cross_l[base + c] * sl[c];
    a0 += vr * gw[c]        + vd * gw[BCH + c]        + vl * gw[2 * BCH + c];
    a1 += vr * gw[3075 + c] + vd * gw[3075 + BCH + c] + vl * gw[3075 + 2 * BCH + c];
    a2 += vr * gw[6150 + c] + vd * gw[6150 + BCH + c] + vl * gw[6150 + 2 * BCH + c];
  }
  a0 = wave_sum(a0); a1 = wave_sum(a1); a2 = wave_sum(a2);
  const float q0 = var_r[row], q1 = sp_d[row], q2 = sp_l[row];
  a0 += gw[3072] * q0 + gw[3073] * q1 + gw[3074] * q2 + gb[0];
  a1 += gw[3075 + 3072] * q0 + gw[3075 + 3073] * q1 + gw[3075 + 3074] * q2 + gb[1];
  a2 += gw[6150 + 3072] * q0 + gw[6150 + 3073] * q1 + gw[6150 + 3074] * q2 + gb[2];
  const float g0 = 1.f / (1.f + __expf(-a0));
  const float g1 = 1.f / (1.f + __expf(-a1));
  const float g2 = 1.f / (1.f + __expf(-a2));
  bf16* f = fused + (long long)row * (3 * BCH);
  for (int c = lane; c < BCH; c += 32) {
    f[c]           = (bf16)(cross_r[base + c] * sr[c] * g0);
    f[BCH + c]     = (bf16)(cross_d[base + c] * sd[c] * g1);
    f[2 * BCH + c] = (bf16)(cross_l[base + c] * sl[c] * g2);
  }
}

// ---------------------------------------------------------------------------
extern "C" void kernel_launch(void* const* d_in, const int* in_sizes, int n_in,
                              void* d_out, int out_size, void* d_ws, size_t ws_size,
                              hipStream_t stream) {
  const float* rgb_f        = (const float*)d_in[0];
  const float* depth_f      = (const float*)d_in[1];
  const float* lidar_f      = (const float*)d_in[2];
  const float* proj_rgb_w   = (const float*)d_in[3];
  const float* proj_rgb_b   = (const float*)d_in[4];
  const float* proj_depth_w = (const float*)d_in[5];
  const float* proj_depth_b = (const float*)d_in[6];
  const float* proj_lidar_w = (const float*)d_in[7];
  const float* proj_lidar_b = (const float*)d_in[8];
  const float* attn_q_w = (const float*)d_in[9];
  const float* attn_q_b = (const float*)d_in[10];
  const float* attn_k_w = (const float*)d_in[11];
  const float* attn_k_b = (const float*)d_in[12];
  const float* attn_v_w = (const float*)d_in[13];
  const float* attn_v_b = (const float*)d_in[14];
  const float* attn_o_w = (const float*)d_in[15];
  const float* attn_o_b = (const float*)d_in[16];
  const float* se_w1 = (const float*)d_in[17];
  const float* se_b1 = (const float*)d_in[18];
  const float* se_w2 = (const float*)d_in[19];
  const float* se_b2 = (const float*)d_in[20];
  const float* gate_w = (const float*)d_in[21];
  const float* gate_b = (const float*)d_in[22];
  const float* fusion_w = (const float*)d_in[23];
  const float* fusion_b = (const float*)d_in[24];
  float* out = (float*)d_out;

  // ---- workspace carve-up (~190 MB, 256B aligned) ----
  char* wp = (char*)d_ws;
  auto alloc = [&](size_t bytes) -> char* {
    char* r = wp;
    wp += (bytes + 255) & ~(size_t)255;
    return r;
  };
  bf16* rgb_sm   = (bf16*)alloc((size_t)BATCH * HWSP * 512 * 2);
  bf16* depth_sm = (bf16*)alloc((size_t)BATCH * HWSP * 256 * 2);
  bf16* lidar_sm = (bf16*)alloc((size_t)BATCH * HWSP * 64 * 2);
  bf16* wp_rgb   = (bf16*)alloc((size_t)BCH * 512 * 2);
  bf16* wp_depth = (bf16*)alloc((size_t)BCH * 256 * 2);
  bf16* wp_lidar = (bf16*)alloc((size_t)BCH * 64 * 2);
  bf16* wq_b = (bf16*)alloc((size_t)BCH * BCH * 2);
  bf16* wk_b = (bf16*)alloc((size_t)BCH * BCH * 2);
  bf16* wv_b = (bf16*)alloc((size_t)BCH * BCH * 2);
  bf16* wo_b = (bf16*)alloc((size_t)BCH * BCH * 2);
  bf16* wf_b = (bf16*)alloc((size_t)3072 * 3072 * 2);
  float* raw = (float*)alloc((size_t)BATCH * HWSP * BCH * 4);
  bf16* r_sm = (bf16*)alloc((size_t)BATCH * HWSP * BCH * 2);
  bf16* d_sm = (bf16*)alloc((size_t)BATCH * HWSP * BCH * 2);
  bf16* l_sm = (bf16*)alloc((size_t)BATCH * HWSP * BCH * 2);
  float* cross_r = (float*)alloc((size_t)BATCH * HWSP * BCH * 4);
  float* cross_d = (float*)alloc((size_t)BATCH * HWSP * BCH * 4);
  float* cross_l = (float*)alloc((size_t)BATCH * HWSP * BCH * 4);
  float* var_r = (float*)alloc((size_t)BATCH * HWSP * 4);
  float* sp_d  = (float*)alloc((size_t)BATCH * HWSP * 4);
  float* sp_l  = (float*)alloc((size_t)BATCH * HWSP * 4);
  bf16* Qcm  = (bf16*)alloc((size_t)BATCH * BCH * HWSP * 2);
  bf16* Kcm  = (bf16*)alloc((size_t)BATCH * BCH * HWSP * 2);
  bf16* Vsm  = (bf16*)alloc((size_t)BATCH * HWSP * BCH * 2);
  bf16* AOsm = (bf16*)alloc((size_t)BATCH * HWSP * BCH * 2);
  float* scores = (float*)alloc((size_t)BATCH * NHEADS * HDIM * HDIM * 4);
  bf16* attn_p  = (bf16*)alloc((size_t)BATCH * NHEADS * HDIM * HDIM * 2);
  float* pool3 = (float*)alloc((size_t)3 * BATCH * BCH * 4);
  float* se3   = (float*)alloc((size_t)3 * BATCH * BCH * 4);
  bf16* fused  = (bf16*)alloc((size_t)BATCH * HWSP * 3 * BCH * 2);
  (void)ws_size; (void)n_in; (void)in_sizes; (void)out_size;

  const dim3 blk(256);
  const long long SBC = (long long)HWSP * BCH;  // 1024*1024

  // ---- precision conversion + layout ----
  to_sm_bf16<<<dim3((BATCH * HWSP * 512 + 255) / 256), dim3(256), 0, stream>>>(rgb_f, rgb_sm, 512, BATCH * HWSP * 512);
  to_sm_bf16<<<dim3((BATCH * HWSP * 256 + 255) / 256), dim3(256), 0, stream>>>(depth_f, depth_sm, 256, BATCH * HWSP * 256);
  to_sm_bf16<<<dim3((BATCH * HWSP * 64 + 255) / 256), dim3(256), 0, stream>>>(lidar_f, lidar_sm, 64, BATCH * HWSP * 64);
  to_bf16<<<dim3((BCH * 512 + 255) / 256), dim3(256), 0, stream>>>(proj_rgb_w, wp_rgb, BCH * 512);
  to_bf16<<<dim3((BCH * 256 + 255) / 256), dim3(256), 0, stream>>>(proj_depth_w, wp_depth, BCH * 256);
  to_bf16<<<dim3((BCH * 64 + 255) / 256), dim3(256), 0, stream>>>(proj_lidar_w, wp_lidar, BCH * 64);
  to_bf16<<<dim3((3072 * 3072 + 255) / 256), dim3(256), 0, stream>>>(fusion_w, wf_b, 3072 * 3072);

  // ---- projections + L2 norm (+ residual init, + quality metrics) ----
  wmma_gemm<false, false, true, false><<<dim3(8, 8, BATCH), blk, 0, stream>>>(
      rgb_sm, wp_rgb, proj_rgb_b, raw, 512, 512, 512, BCH,
      (long long)HWSP * 512, 0, 0, 0, SBC, 0, 1);
  l2norm_kernel<<<dim3(BATCH * HWSP / 8), dim3(256), 0, stream>>>(raw, r_sm, cross_r, var_r, 0);

  wmma_gemm<false, false, true, false><<<dim3(8, 8, BATCH), blk, 0, stream>>>(
      depth_sm, wp_depth, proj_depth_b, raw, 256, 256, 256, BCH,
      (long long)HWSP * 256, 0, 0, 0, SBC, 0, 1);
  l2norm_kernel<<<dim3(BATCH * HWSP / 8), dim3(256), 0, stream>>>(raw, d_sm, cross_d, sp_d, 1);

  wmma_gemm<false, false, true, false><<<dim3(8, 8, BATCH), blk, 0, stream>>>(
      lidar_sm, wp_lidar, proj_lidar_b, raw, 64, 64, 64, BCH,
      (long long)HWSP * 64, 0, 0, 0, SBC, 0, 1);
  l2norm_kernel<<<dim3(BATCH * HWSP / 8), dim3(256), 0, stream>>>(raw, l_sm, cross_l, sp_l, 1);

  // ---- 6 cross-attention blocks ----
  const bf16* qf[6]  = {r_sm, r_sm, d_sm, d_sm, l_sm, l_sm};
  const bf16* kvf[6] = {d_sm, l_sm, r_sm, l_sm, r_sm, d_sm};
  float* tgt[6]      = {cross_r, cross_r, cross_d, cross_d, cross_l, cross_l};

  for (int i = 0; i < 6; ++i) {
    const size_t wofs = (size_t)i * BCH * BCH;
    to_bf16<<<dim3(4096), dim3(256), 0, stream>>>(attn_q_w + wofs, wq_b, BCH * BCH);
    to_bf16<<<dim3(4096), dim3(256), 0, stream>>>(attn_k_w + wofs, wk_b, BCH * BCH);
    to_bf16<<<dim3(4096), dim3(256), 0, stream>>>(attn_v_w + wofs, wv_b, BCH * BCH);
    to_bf16<<<dim3(4096), dim3(256), 0, stream>>>(attn_o_w + wofs, wo_b, BCH * BCH);

    // Q,K -> channel-major bf16 (transposed store); V -> spatial-major bf16
    wmma_gemm<true, true, true, false><<<dim3(8, 8, BATCH), blk, 0, stream>>>(
        qf[i], wq_b, attn_q_b + i * BCH, Qcm, BCH, BCH, BCH, HWSP,
        SBC, 0, 0, 0, SBC, 0, 1);
    wmma_gemm<true, true, true, false><<<dim3(8, 8, BATCH), blk, 0, stream>>>(
        kvf[i], wk_b, attn_k_b + i * BCH, Kcm, BCH, BCH, BCH, HWSP,
        SBC, 0, 0, 0, SBC, 0, 1);
    wmma_gemm<true, false, true, false><<<dim3(8, 8, BATCH), blk, 0, stream>>>(
        kvf[i], wv_b, attn_v_b + i * BCH, Vsm, BCH, BCH, BCH, BCH,
        SBC, 0, 0, 0, SBC, 0, 1);

    // scores[b,h] = Q_h (128 x 1024) . K_h^T  -> f32 [b,h,128,128]
    wmma_gemm<false, false, false, false><<<dim3(1, 1, BATCH * NHEADS), blk, 0, stream>>>(
        Qcm, Kcm, nullptr, scores, HWSP, HWSP, HWSP, HDIM,
        SBC, (long long)HDIM * HWSP, SBC, (long long)HDIM * HWSP,
        (long long)NHEADS * HDIM * HDIM, (long long)HDIM * HDIM, NHEADS);

    softmax_kernel<<<dim3(BATCH * NHEADS * HDIM / 8), dim3(256), 0, stream>>>(scores, attn_p);

    // out[b,h] = attn (128x128) . V_h -> spatial-major bf16 (transposed store)
    wmma_gemm<true, true, false, false><<<dim3(8, 1, BATCH * NHEADS), blk, 0, stream>>>(
        attn_p, Vsm, nullptr, AOsm, HDIM, HDIM, BCH, BCH,
        (long long)NHEADS * HDIM * HDIM, (long long)HDIM * HDIM,
        SBC, HDIM, SBC, HDIM, NHEADS);

    // O projection, accumulated into the cross residual buffer
    wmma_gemm<false, false, true, true><<<dim3(8, 8, BATCH), blk, 0, stream>>>(
        AOsm, wo_b, attn_o_b + i * BCH, tgt[i], BCH, BCH, BCH, BCH,
        SBC, 0, 0, 0, SBC, 0, 1);
  }

  // ---- SE gates ----
  const float* crs[3] = {cross_r, cross_d, cross_l};
  for (int m = 0; m < 3; ++m) {
    pool_kernel<<<dim3(16), dim3(256), 0, stream>>>(crs[m], pool3 + m * BATCH * BCH);
    se_kernel<<<dim3(BATCH), dim3(256), 0, stream>>>(
        pool3 + m * BATCH * BCH, se_w1 + (size_t)m * 64 * BCH, se_b1 + m * 64,
        se_w2 + (size_t)m * BCH * 64, se_b2 + m * BCH, se3 + m * BATCH * BCH);
  }

  // ---- gating + fused feature assembly (bf16) ----
  gate_fused_kernel<<<dim3(BATCH * HWSP / 8), dim3(256), 0, stream>>>(
      cross_r, cross_d, cross_l,
      se3, se3 + BATCH * BCH, se3 + 2 * BATCH * BCH,
      var_r, sp_d, sp_l, gate_w, gate_b, fused);

  // ---- final fusion GEMM -> d_out [B, 3072, 32, 32] (channel-major f32) ----
  wmma_gemm<false, true, true, false><<<dim3(24, 8, BATCH), blk, 0, stream>>>(
      fused, wf_b, fusion_b, out, 3072, 3072, 3072, HWSP,
      (long long)HWSP * 3072, 0, 0, 0, (long long)3072 * HWSP, 0, 1);
}
